// GQABlock_49495203119702
// MI455X (gfx1250) — matrix operand order repaired and checked by reading, not compiled, not run on hardware
//
#include <hip/hip_runtime.h>
#include <hip/hip_bf16.h>
#include <math.h>

typedef __attribute__((ext_vector_type(16))) _Float16 v16h;
typedef __attribute__((ext_vector_type(8)))  float    v8f;

#define B_   2
#define L_   2048
#define D_   2048
#define HQ_  32
#define HKV_ 8
#define HD_  64

// ---------------------------------------------------------------------------
// WMMA fragment helpers (gfx1250, wave32).  ISA 7.12.2 layouts:
//  A (16x32 f16): lane row m = lane&15; half h = lane>>4;
//                 pair j holds K = ((j<4)?0:16) + 8h + 2(j&3)
//  B (32x16 f16): lane col n = lane&15; pair j holds K = ((j<4)?0:16) + 8h + 2(j&3)
//  C/D (16x16 f32): lane col n = lane&15; VGPR r holds row m = r + 8h
// ---------------------------------------------------------------------------
__device__ __forceinline__ int a_koff(int j, int h) {
  return ((j < 4) ? 0 : 16) + h * 8 + (j & 3) * 2;
}
__device__ __forceinline__ int b_koff(int j, int h) {
  return ((j < 4) ? 0 : 16) + h * 8 + (j & 3) * 2;
}

// A fragment, K contiguous at rowptr[k] (row fixed per lane by caller)
__device__ __forceinline__ v16h load_a_frag(const _Float16* rowptr, int h) {
  v16h a;
#pragma unroll
  for (int j = 0; j < 8; ++j) {
    int k = a_koff(j, h);
    a[2 * j]     = rowptr[k];
    a[2 * j + 1] = rowptr[k + 1];
  }
  return a;
}

// B fragment from a transposed operand: element (k,n) = src_row_n[k], K contiguous
__device__ __forceinline__ v16h load_bt_frag(const _Float16* colptr, int h) {
  v16h b;
#pragma unroll
  for (int j = 0; j < 8; ++j) {
    int k = b_koff(j, h);
    b[2 * j]     = colptr[k];
    b[2 * j + 1] = colptr[k + 1];
  }
  return b;
}

// B fragment from a row-major K x N operand: element (k,n) = base[k*ld] (col fixed)
__device__ __forceinline__ v16h load_bn_frag(const _Float16* base, int ld, int h) {
  v16h b;
#pragma unroll
  for (int j = 0; j < 8; ++j) {
    int k = b_koff(j, h);
    b[2 * j]     = base[(size_t)k * ld];
    b[2 * j + 1] = base[(size_t)(k + 1) * ld];
  }
  return b;
}

__device__ __forceinline__ v8f wmma_f16(v16h a, v16h b, v8f c) {
  return __builtin_amdgcn_wmma_f32_16x16x32_f16(false, a, false, b, (short)0, c,
                                                false, false);
}

// ---------------------------------------------------------------------------
// f32 -> f16 conversion
// ---------------------------------------------------------------------------
__global__ void __launch_bounds__(256) f32_to_f16_kernel(const float* __restrict__ in,
                                                         _Float16* __restrict__ out,
                                                         size_t n) {
  size_t i = (size_t)blockIdx.x * blockDim.x + threadIdx.x;
  if (i < n) out[i] = (_Float16)in[i];
}

// ---------------------------------------------------------------------------
// Y[b][head][l][d] = (X @ W^T), X:(B*L, D) f16, W:(N, D) f16, f16 output.
// One wave computes a 64x64 tile: 4 A-frags x 4 B-frags -> 16 WMMAs per
// 32-wide K step (512 B of L2 traffic per WMMA).
// ---------------------------------------------------------------------------
__global__ void __launch_bounds__(256) qkv_gemm_kernel(const _Float16* __restrict__ X,
                                                       const _Float16* __restrict__ W,
                                                       _Float16* __restrict__ Y,
                                                       int N, int H) {
  const int lane = threadIdx.x & 31;
  const int h    = lane >> 4;
  const int n16  = lane & 15;
  const int wid  = (int)((blockIdx.x * blockDim.x + threadIdx.x) >> 5);
  const int ntiles = N >> 6;
  const int m0 = (wid / ntiles) << 6;
  const int n0 = (wid % ntiles) << 6;

  v8f acc[4][4] = {};
  const _Float16* xr[4];
  const _Float16* wr[4];
#pragma unroll
  for (int i = 0; i < 4; ++i) {
    xr[i] = X + (size_t)(m0 + i * 16 + n16) * D_;
    wr[i] = W + (size_t)(n0 + i * 16 + n16) * D_;
  }

  for (int k0 = 0; k0 < D_; k0 += 32) {
    v16h a[4], bb[4];
#pragma unroll
    for (int i = 0; i < 4; ++i) a[i] = load_a_frag(xr[i] + k0, h);
#pragma unroll
    for (int i = 0; i < 4; ++i) bb[i] = load_bt_frag(wr[i] + k0, h);
    if (k0 + 128 < D_) {
#pragma unroll
      for (int i = 0; i < 4; ++i) {
        __builtin_prefetch(xr[i] + k0 + 128, 0, 3);   // global_prefetch_b8
        __builtin_prefetch(wr[i] + k0 + 128, 0, 3);
      }
    }
#pragma unroll
    for (int mi = 0; mi < 4; ++mi)
#pragma unroll
      for (int ni = 0; ni < 4; ++ni)
        acc[mi][ni] = wmma_f16(a[mi], bb[ni], acc[mi][ni]);
  }

#pragma unroll
  for (int mi = 0; mi < 4; ++mi)
#pragma unroll
    for (int ni = 0; ni < 4; ++ni) {
      int col = n0 + ni * 16 + n16;
      int hh = col >> 6;        // head
      int d  = col & 63;        // dim within head
#pragma unroll
      for (int r = 0; r < 8; ++r) {
        int m = m0 + mi * 16 + r + 8 * h;
        int b = m >> 11;        // / L_
        int l = m & (L_ - 1);
        Y[(((size_t)b * H + hh) * L_ + l) * HD_ + d] = (_Float16)acc[mi][ni][r];
      }
    }
}

// ---------------------------------------------------------------------------
// In-place RoPE on [B][H][L][HD] f16 (pairs (2i, 2i+1) along head dim)
// ---------------------------------------------------------------------------
__global__ void __launch_bounds__(256) rope_kernel(_Float16* __restrict__ q, size_t total) {
  size_t idx = (size_t)blockIdx.x * blockDim.x + threadIdx.x;
  if (idx >= total) return;
  int i = (int)(idx & 31);            // pair index, HD/2 = 32
  size_t t = idx >> 5;
  int l = (int)(t & (L_ - 1));
  size_t bh = t >> 11;
  // inv = 10000^(-(2i)/64)
  float inv = __expf(-((float)(2 * i) * (1.0f / 64.0f)) * 9.210340371976184f);
  float f = (float)l * inv;
  float c = cosf(f), s = sinf(f);
  _Float16* p = q + (bh * L_ + (size_t)l) * HD_ + 2 * i;
  float x1 = (float)p[0], x2 = (float)p[1];
  p[0] = (_Float16)(x1 * c - x2 * s);
  p[1] = (_Float16)(x1 * s + x2 * c);
}

// ---------------------------------------------------------------------------
// Flash attention: one wave owns 16 query rows of one (b, hq) pair.
// Causal loop over 32-key blocks; QK^T and PV on WMMA; online softmax with
// 16-lane shuffle reductions; P transposed C-layout -> A-layout through LDS.
// ---------------------------------------------------------------------------
#define ATTN_WPB 4
__global__ void __launch_bounds__(ATTN_WPB * 32) attn_kernel(const _Float16* __restrict__ qh,
                                                             const _Float16* __restrict__ kh,
                                                             const _Float16* __restrict__ vh,
                                                             _Float16* __restrict__ oh) {
  __shared__ _Float16 plds[ATTN_WPB][16 * 32];

  const int lane = threadIdx.x & 31;
  const int warp = threadIdx.x >> 5;
  const int h    = lane >> 4;
  const int n16  = lane & 15;

  const int wg  = blockIdx.x * ATTN_WPB + warp;
  const int qt  = wg & 127;             // L_/16 query tiles
  const int hq  = (wg >> 7) & 31;
  const int b   = wg >> 12;
  const int hkv = hq >> 2;              // G = 4
  const int q0  = qt << 4;

  // Q A-fragments (d 0..31 and 32..63), lane row = q0 + n16
  const _Float16* qrow = qh + (((size_t)b * HQ_ + hq) * L_ + q0 + n16) * HD_;
  v16h aq0 = load_a_frag(qrow, h);
  v16h aq1 = load_a_frag(qrow + 32, h);

  const _Float16* kbase = kh + ((size_t)b * HKV_ + hkv) * L_ * HD_;
  const _Float16* vbase = vh + ((size_t)b * HKV_ + hkv) * L_ * HD_;

  float mrow[8], lrow[8];
#pragma unroll
  for (int r = 0; r < 8; ++r) { mrow[r] = -INFINITY; lrow[r] = 0.0f; }
  v8f o[4] = {};

  _Float16* pl = plds[warp];
  const int nkb = (q0 + 47) >> 5;       // causal: keys up to q0+15

  for (int kb = 0; kb < nkb; ++kb) {
    const int kstart = kb << 5;

    // K^T B-fragments: two 16-key groups x two 32-wide d-chunks
    const _Float16* kr0 = kbase + (size_t)(kstart + n16) * HD_;
    const _Float16* kr1 = kbase + (size_t)(kstart + 16 + n16) * HD_;
    v16h b00 = load_bt_frag(kr0, h);        // keys 0..15, d 0..31
    v16h b01 = load_bt_frag(kr0 + 32, h);   // keys 0..15, d 32..63
    v16h b10 = load_bt_frag(kr1, h);        // keys 16..31, d 0..31
    v16h b11 = load_bt_frag(kr1 + 32, h);   // keys 16..31, d 32..63

    v8f s0 = {}, s1 = {};
    s0 = wmma_f16(aq0, b00, s0);
    s0 = wmma_f16(aq1, b01, s0);
    s1 = wmma_f16(aq0, b10, s1);
    s1 = wmma_f16(aq1, b11, s1);

    // scale + causal mask + online softmax
    float p0[8], p1[8], scl[8];
#pragma unroll
    for (int r = 0; r < 8; ++r) {
      int row  = q0 + r + 8 * h;
      int key0 = kstart + n16;
      int key1 = kstart + 16 + n16;
      float v0 = (key0 <= row) ? s0[r] * 0.125f : -INFINITY;
      float v1 = (key1 <= row) ? s1[r] * 0.125f : -INFINITY;

      float mx = fmaxf(v0, v1);
#pragma unroll
      for (int d = 1; d <= 8; d <<= 1)
        mx = fmaxf(mx, __shfl_xor(mx, d, 32));   // 16-lane reduce, stays in half

      float mnew = fmaxf(mrow[r], mx);
      scl[r] = __expf(mrow[r] - mnew);
      p0[r] = __expf(v0 - mnew);
      p1[r] = __expf(v1 - mnew);

      float rs = p0[r] + p1[r];
#pragma unroll
      for (int d = 1; d <= 8; d <<= 1)
        rs += __shfl_xor(rs, d, 32);

      lrow[r] = lrow[r] * scl[r] + rs;
      mrow[r] = mnew;
    }

    // rescale running O
#pragma unroll
    for (int nc = 0; nc < 4; ++nc)
#pragma unroll
      for (int r = 0; r < 8; ++r) o[nc][r] *= scl[r];

    // P (C-layout) -> LDS -> A-layout fragment
#pragma unroll
    for (int r = 0; r < 8; ++r) {
      pl[(r + 8 * h) * 32 + n16]      = (_Float16)p0[r];
      pl[(r + 8 * h) * 32 + 16 + n16] = (_Float16)p1[r];
    }
    asm volatile("s_wait_dscnt 0" ::: "memory");
    v16h pa = load_a_frag(pl + (size_t)n16 * 32, h);

    // O += P x V  (K-dim = 32 keys of this block)
    const _Float16* vblk = vbase + (size_t)kstart * HD_;
#pragma unroll
    for (int nc = 0; nc < 4; ++nc) {
      v16h bv = load_bn_frag(vblk + nc * 16 + n16, HD_, h);
      o[nc] = wmma_f16(pa, bv, o[nc]);
    }
  }

  // epilogue: normalize and store f16 into (B*L, HQ*HD)
  float invl[8];
#pragma unroll
  for (int r = 0; r < 8; ++r) invl[r] = 1.0f / lrow[r];
#pragma unroll
  for (int nc = 0; nc < 4; ++nc) {
    int col = hq * HD_ + nc * 16 + n16;
#pragma unroll
    for (int r = 0; r < 8; ++r) {
      int l = q0 + r + 8 * h;
      oh[((size_t)b * L_ + l) * (HQ_ * HD_) + col] = (_Float16)(o[nc][r] * invl[r]);
    }
  }
}

// ---------------------------------------------------------------------------
// Final projection: Y(f32, B*L x D) = X(f16, B*L x D) @ W(f16, D x D)^T
// 64x64 wave tile, same blocking as qkv_gemm_kernel.
// ---------------------------------------------------------------------------
__global__ void __launch_bounds__(256) out_gemm_kernel(const _Float16* __restrict__ X,
                                                       const _Float16* __restrict__ W,
                                                       float* __restrict__ Y) {
  const int lane = threadIdx.x & 31;
  const int h    = lane >> 4;
  const int n16  = lane & 15;
  const int wid  = (int)((blockIdx.x * blockDim.x + threadIdx.x) >> 5);
  const int ntiles = D_ >> 6;
  const int m0 = (wid / ntiles) << 6;
  const int n0 = (wid % ntiles) << 6;

  v8f acc[4][4] = {};
  const _Float16* xr[4];
  const _Float16* wr[4];
#pragma unroll
  for (int i = 0; i < 4; ++i) {
    xr[i] = X + (size_t)(m0 + i * 16 + n16) * D_;
    wr[i] = W + (size_t)(n0 + i * 16 + n16) * D_;
  }

  for (int k0 = 0; k0 < D_; k0 += 32) {
    v16h a[4], bb[4];
#pragma unroll
    for (int i = 0; i < 4; ++i) a[i] = load_a_frag(xr[i] + k0, h);
#pragma unroll
    for (int i = 0; i < 4; ++i) bb[i] = load_bt_frag(wr[i] + k0, h);
    if (k0 + 128 < D_) {
#pragma unroll
      for (int i = 0; i < 4; ++i) {
        __builtin_prefetch(xr[i] + k0 + 128, 0, 3);
        __builtin_prefetch(wr[i] + k0 + 128, 0, 3);
      }
    }
#pragma unroll
    for (int mi = 0; mi < 4; ++mi)
#pragma unroll
      for (int ni = 0; ni < 4; ++ni)
        acc[mi][ni] = wmma_f16(a[mi], bb[ni], acc[mi][ni]);
  }

#pragma unroll
  for (int mi = 0; mi < 4; ++mi)
#pragma unroll
    for (int ni = 0; ni < 4; ++ni) {
      int col = n0 + ni * 16 + n16;
#pragma unroll
      for (int r = 0; r < 8; ++r) {
        Y[(size_t)(m0 + mi * 16 + r + 8 * h) * D_ + col] = acc[mi][ni][r];
      }
    }
}

// ---------------------------------------------------------------------------
extern "C" void kernel_launch(void* const* d_in, const int* in_sizes, int n_in,
                              void* d_out, int out_size, void* d_ws, size_t ws_size,
                              hipStream_t stream) {
  (void)in_sizes; (void)n_in; (void)out_size; (void)ws_size;
  const float* x  = (const float*)d_in[0];
  const float* Wq = (const float*)d_in[1];
  const float* Wk = (const float*)d_in[2];
  const float* Wv = (const float*)d_in[3];
  const float* Wo = (const float*)d_in[4];

  const size_t n_x  = (size_t)B_ * L_ * D_;         // 8.39M
  const size_t n_wq = (size_t)HQ_ * HD_ * D_;       // 4.19M
  const size_t n_wk = (size_t)HKV_ * HD_ * D_;      // 1.05M
  const size_t n_q  = (size_t)B_ * HQ_ * L_ * HD_;  // 8.39M
  const size_t n_k  = (size_t)B_ * HKV_ * L_ * HD_; // 2.10M

  _Float16* p = (_Float16*)d_ws;
  _Float16* xh   = p; p += n_x;
  _Float16* wq16 = p; p += n_wq;
  _Float16* wk16 = p; p += n_wk;
  _Float16* wv16 = p; p += n_wk;
  _Float16* wo16 = p; p += n_wq;
  _Float16* qh   = p; p += n_q;
  _Float16* kh   = p; p += n_k;
  _Float16* vh   = p; p += n_k;
  _Float16* oh   = p; p += n_q;

  auto cvt = [&](const float* src, _Float16* dst, size_t n) {
    f32_to_f16_kernel<<<(unsigned)((n + 255) / 256), 256, 0, stream>>>(src, dst, n);
  };
  cvt(x,  xh,   n_x);
  cvt(Wq, wq16, n_wq);
  cvt(Wk, wk16, n_wk);
  cvt(Wv, wv16, n_wk);
  cvt(Wo, wo16, n_wq);

  const int M = B_ * L_;  // 4096
  // QKV projections: waves = (M/64)*(N/64), 8 waves per 256-thread block
  qkv_gemm_kernel<<<(M / 64) * (2048 / 64) / 8, 256, 0, stream>>>(xh, wq16, qh, 2048, HQ_);
  qkv_gemm_kernel<<<(M / 64) * (512 / 64) / 8, 256, 0, stream>>>(xh, wk16, kh, 512, HKV_);
  qkv_gemm_kernel<<<(M / 64) * (512 / 64) / 8, 256, 0, stream>>>(xh, wv16, vh, 512, HKV_);

  // RoPE (in place; buffers fully rewritten by the GEMMs each call)
  rope_kernel<<<(unsigned)((n_q / 2 + 255) / 256), 256, 0, stream>>>(qh, n_q / 2);
  rope_kernel<<<(unsigned)((n_k / 2 + 255) / 256), 256, 0, stream>>>(kh, n_k / 2);

  // Attention: B*HQ*(L/16) = 8192 waves, 4 waves per block
  attn_kernel<<<(B_ * HQ_ * (L_ / 16)) / ATTN_WPB, ATTN_WPB * 32, 0, stream>>>(qh, kh, vh, oh);

  // Output projection -> f32 d_out
  out_gemm_kernel<<<(M / 64) * (2048 / 64) / 8, 256, 0, stream>>>(oh, wo16, (float*)d_out);
}
